// MambaEncoderPatchMergingBlock_64630667870743
// MI455X (gfx1250) — matrix operand order repaired and checked by reading, not compiled, not run on hardware
//
#include <hip/hip_runtime.h>
#include <hip/hip_bf16.h>

// ---------------------------------------------------------------------------
// Shapes / constants
// ---------------------------------------------------------------------------
#define BB      8
#define TT      8
#define WW      32
#define DIMC    192
#define D_MODEL 384
#define D_INNER 768
#define D_STATE 16
#define DT_RANK 24
#define DT_PAD  32            // DT_RANK padded to 32 for WMMA K
#define XP_N    56            // DT_RANK + 2*D_STATE
#define L_FULL  2049          // 1 cls + 8*256 merged tokens
#define M_FULL  (BB * L_FULL) // 16392
#define M_MERGE (BB * 2048)   // 16384
#define EPSV    1e-5f
#define NCH     8             // scan chunks
#define CHL     257           // ceil(2049/8)
#define NT      4             // N-tiles (of 16) per wave in GEMM
#define BN_RB   64            // batch-norm row blocks
#define BN_RPB  257           // rows per bn block (64*257 >= 16392)

typedef __bf16 bf16;
typedef bf16  bf16x16 __attribute__((ext_vector_type(16)));
typedef float f32x8   __attribute__((ext_vector_type(8)));
typedef unsigned short u16x8 __attribute__((ext_vector_type(8)));

union ABFrag { bf16x16 v; u16x8 h[2]; };

__device__ __forceinline__ unsigned short f2bf(float f) {
    union { float f; unsigned u; } v; v.f = f;
    unsigned r = v.u + 0x7FFFu + ((v.u >> 16) & 1u);   // round-to-nearest-even
    return (unsigned short)(r >> 16);
}

// ---------------------------------------------------------------------------
// bf16 WMMA GEMM:  C[M,N] (f32) = A[M,K] @ W[N,K]^T
// Barrier-free: all 10 fragment loads of a K-step are issued as one group
// before the 4 WMMAs; the backend emits partial s_wait_loadcnt so wmma[t]
// overlaps with the still-outstanding loads of b[t+1..] (loads retire in
// order). Single-buffered fragments keep VGPR pressure low (no spills).
// Each wave: 16 rows x 64 cols (4 accumulators share one A fragment).
// mode bit0: C = softplus(C + bias[n]);  mode bit1: also store bf16 copy.
// ---------------------------------------------------------------------------
__global__ void k_gemm_bf16(const unsigned short* __restrict__ A,
                            const unsigned short* __restrict__ W,
                            float* __restrict__ C,
                            int M, int N, int K,
                            const float* __restrict__ bias,
                            unsigned short* __restrict__ Cbf,
                            int mode)
{
    const int tid  = threadIdx.x;
    const int wave = tid >> 5;
    const int lane = tid & 31;
    const int half = lane >> 4;     // 0..1
    const int r    = lane & 15;     // 0..15

    const int n0 = blockIdx.x * (16 * NT);
    const int m0 = blockIdx.y * 128 + wave * 16;

    int arow = m0 + r;
    if (arow >= M) arow = M - 1;            // clamp loads; stores guarded
    const unsigned short* Ap = A + (size_t)arow * K + 8 * half;

    const unsigned short* Bp[NT];
#pragma unroll
    for (int t = 0; t < NT; ++t) {
        int wn = n0 + 16 * t + r;
        if (wn >= N) wn = N - 1;
        Bp[t] = W + (size_t)wn * K + 8 * half;
    }

    f32x8 acc[NT];
#pragma unroll
    for (int t = 0; t < NT; ++t) acc[t] = (f32x8){0.f,0.f,0.f,0.f,0.f,0.f,0.f,0.f};

    for (int k0 = 0; k0 < K; k0 += 32) {
        // A fragment (ISA 16-bit A 16x32 layout: half -> K {0..7,16..23} / {8..15,24..31})
        ABFrag a;
        a.h[0] = *(const u16x8*)(Ap + k0);
        a.h[1] = *(const u16x8*)(Ap + k0 + 16);
        // B fragments: lane half selects K {0..7,16..23} / {8..15,24..31}, column = r
        ABFrag b[NT];
#pragma unroll
        for (int t = 0; t < NT; ++t) {
            b[t].h[0] = *(const u16x8*)(Bp[t] + k0);
            b[t].h[1] = *(const u16x8*)(Bp[t] + k0 + 16);
        }
#pragma unroll
        for (int t = 0; t < NT; ++t) {
            acc[t] = __builtin_amdgcn_wmma_f32_16x16x32_bf16(
                         false, a.v, false, b[t].v, (short)0, acc[t], false, false);
        }
    }

#pragma unroll
    for (int t = 0; t < NT; ++t) {
        const int col = n0 + 16 * t + r;
        if (col >= N) continue;
#pragma unroll
        for (int i = 0; i < 8; ++i) {
            int row = m0 + i + 8 * half;    // D layout: VGPR i -> M=i / M=8+i
            if (row < M) {
                float v = acc[t][i];
                if (mode & 1) {
                    v += bias[col];
                    v = (v > 20.f) ? v : log1pf(__expf(v));   // softplus
                }
                C[(size_t)row * N + col] = v;
                if (mode & 2) Cbf[(size_t)row * N + col] = f2bf(v);
            }
        }
    }
}

// ---------------------------------------------------------------------------
// Patch merge + LayerNorm -> bf16 tokens [B*2048, 768]. One wave per token.
// ---------------------------------------------------------------------------
__global__ void k_patch_merge_ln(const float* __restrict__ x,
                                 const float* __restrict__ g,
                                 const float* __restrict__ bta,
                                 unsigned short* __restrict__ m_bf)
{
    int gwave = (int)((blockIdx.x * (size_t)blockDim.x + threadIdx.x) >> 5);
    int lane  = threadIdx.x & 31;
    if (gwave >= BB * 2048) return;
    int b = gwave >> 11;
    int l = gwave & 2047;
    int t = l >> 8, p = l & 255;
    int i2 = p >> 4, j2 = p & 15;

    float v[24];
    float s = 0.f;
#pragma unroll
    for (int j = 0; j < 24; ++j) {
        int c = lane + 32 * j;
        int q = c / DIMC, cc = c % DIMC;
        int rr  = 2 * i2 + (q & 1);
        int col = 2 * j2 + (q >> 1);
        size_t idx = ((size_t)b * (1 + TT * 1024) + 1 +
                      (size_t)t * 1024 + rr * WW + col) * DIMC + cc;
        v[j] = x[idx];
        s += v[j];
    }
#pragma unroll
    for (int off = 16; off; off >>= 1) s += __shfl_xor(s, off, 32);
    float mu = s * (1.f / 768.f);
    float s2 = 0.f;
#pragma unroll
    for (int j = 0; j < 24; ++j) { float d = v[j] - mu; s2 += d * d; }
#pragma unroll
    for (int off = 16; off; off >>= 1) s2 += __shfl_xor(s2, off, 32);
    float rs = rsqrtf(s2 * (1.f / 768.f) + EPSV);

    size_t ob = (size_t)(b * 2048 + l) * 768;
#pragma unroll
    for (int j = 0; j < 24; ++j) {
        int c = lane + 32 * j;
        m_bf[ob + c] = f2bf((v[j] - mu) * rs * g[c] + bta[c]);
    }
}

// ---------------------------------------------------------------------------
// Assemble h[B,2049,384]: row 0 = cls@W_cls^T + b_cls, rows 1.. = m_out copy.
// ---------------------------------------------------------------------------
__global__ void k_assemble_h(const float* __restrict__ m_out,
                             const float* __restrict__ x,
                             const float* __restrict__ Wcls,
                             const float* __restrict__ bcls,
                             float* __restrict__ h,
                             unsigned short* __restrict__ h_bf)
{
    size_t idx = blockIdx.x * (size_t)blockDim.x + threadIdx.x;
    if (idx >= (size_t)M_FULL * D_MODEL) return;
    int n = (int)(idx % D_MODEL);
    size_t row = idx / D_MODEL;
    int b = (int)(row / L_FULL);
    int l = (int)(row % L_FULL);
    float v;
    if (l == 0) {
        const float* cls = x + (size_t)b * (1 + TT * 1024) * DIMC;
        const float* w   = Wcls + (size_t)n * DIMC;
        float acc = bcls[n];
        for (int k = 0; k < DIMC; ++k) acc += cls[k] * w[k];
        v = acc;
    } else {
        v = m_out[((size_t)b * 2048 + (l - 1)) * D_MODEL + n];
    }
    h[idx] = v;
    h_bf[idx] = f2bf(v);
}

// ---------------------------------------------------------------------------
// Depthwise causal conv (k=4) + bias + SiLU. xz layout [M,1536], xs = cols 0..767.
// ---------------------------------------------------------------------------
__global__ void k_conv_silu(const float* __restrict__ xz,
                            const float* __restrict__ cw,   // [768][4]
                            const float* __restrict__ cb,   // [768]
                            float* __restrict__ xs,
                            unsigned short* __restrict__ xs_bf)
{
    size_t idx = blockIdx.x * (size_t)blockDim.x + threadIdx.x;
    if (idx >= (size_t)M_FULL * D_INNER) return;
    int c = (int)(idx % D_INNER);
    size_t row = idx / D_INNER;
    int l = (int)(row % L_FULL);
    float acc = cb[c];
#pragma unroll
    for (int j = 0; j < 4; ++j) {
        int li = l - 3 + j;
        if (li >= 0)
            acc += cw[c * 4 + j] * xz[(row - 3 + j) * 1536 + c];
    }
    float s = acc / (1.f + __expf(-acc));
    xs[idx]    = s;
    xs_bf[idx] = f2bf(s);
}

// dt part of x_dbl -> bf16, K padded 24->32 with zeros
__global__ void k_dt_pad(const float* __restrict__ xdbl,
                         unsigned short* __restrict__ dt_bf)
{
    size_t idx = blockIdx.x * (size_t)blockDim.x + threadIdx.x;
    if (idx >= (size_t)M_FULL * DT_PAD) return;
    int c = (int)(idx & (DT_PAD - 1));
    size_t row = idx >> 5;
    dt_bf[idx] = (c < DT_RANK) ? f2bf(xdbl[row * XP_N + c]) : (unsigned short)0;
}

// W_dt [2][768][24] -> bf16 [2][768][32] zero-padded
__global__ void k_wdt_pad(const float* __restrict__ Wdt,
                          unsigned short* __restrict__ Wdt_bf)
{
    size_t idx = blockIdx.x * (size_t)blockDim.x + threadIdx.x;
    if (idx >= (size_t)2 * D_INNER * DT_PAD) return;
    int c = (int)(idx & (DT_PAD - 1));
    size_t row = idx >> 5;
    Wdt_bf[idx] = (c < DT_RANK) ? f2bf(Wdt[row * DT_RANK + c]) : (unsigned short)0;
}

__global__ void k_f2bf(const float* __restrict__ src,
                       unsigned short* __restrict__ dst, int n)
{
    for (int i = blockIdx.x * blockDim.x + threadIdx.x; i < n;
         i += gridDim.x * blockDim.x)
        dst[i] = f2bf(src[i]);
}

__global__ void k_aneg(const float* __restrict__ alog,
                       float* __restrict__ aneg, int n)
{
    int i = blockIdx.x * blockDim.x + threadIdx.x;
    if (i < n) aneg[i] = -__expf(alog[i]);
}

// ---------------------------------------------------------------------------
// Chunked selective scan.
//   h_t = exp(A*sum(delta))∘h_init + h_local_t   (diagonal SSM prefix split)
// Phase 1: per-(b,d,chunk) local scan from zero state. 49152 lanes.
// Phase 2: 8-step sequential chunk fix-up (per (b,d)).
// Phase 3: add correction, gate with SiLU(z), emit bf16 for W_out GEMM.
// ---------------------------------------------------------------------------
__global__ void k_scan1(const float* __restrict__ xs,     // [M,768]
                        const float* __restrict__ delta,  // [M,768]
                        const float* __restrict__ xdbl,   // [M,56] (B@24,C@40)
                        const float* __restrict__ aneg,   // [768,16]
                        const float* __restrict__ dskip,  // [768]
                        float* __restrict__ ylocal,       // [M,768]
                        float* __restrict__ hend,         // [B][NCH][16][768]
                        float* __restrict__ dsumb)        // [B][NCH][768]
{
    int gid = blockIdx.x * blockDim.x + threadIdx.x;
    if (gid >= BB * D_INNER * NCH) return;
    int d  = gid % D_INNER;
    int b  = (gid / D_INNER) % BB;
    int ch = gid / (D_INNER * BB);

    float Ar[D_STATE], st[D_STATE];
#pragma unroll
    for (int n = 0; n < D_STATE; ++n) { Ar[n] = aneg[d * D_STATE + n]; st[n] = 0.f; }
    const float Dv = dskip[d];

    const int l0 = ch * CHL;
    const int l1 = (l0 + CHL < L_FULL) ? (l0 + CHL) : L_FULL;
    size_t base = (size_t)b * L_FULL;
    float dsum = 0.f;
    for (int l = l0; l < l1; ++l) {
        size_t row = base + l;
        if (l + 4 < l1) {
            __builtin_prefetch(&xs[(row + 4) * D_INNER + d], 0, 1);
            __builtin_prefetch(&delta[(row + 4) * D_INNER + d], 0, 1);
        }
        float u  = xs[row * D_INNER + d];
        float dl = delta[row * D_INNER + d];
        const float* bc = xdbl + row * XP_N;
        dsum += dl;
        float y = u * Dv;
#pragma unroll
        for (int n = 0; n < D_STATE; ++n) {
            float dA = __expf(dl * Ar[n]);
            st[n] = st[n] * dA + (dl * bc[DT_RANK + n]) * u;
            y += st[n] * bc[DT_RANK + D_STATE + n];
        }
        ylocal[row * D_INNER + d] = y;
    }
    size_t hb = ((size_t)(b * NCH + ch)) * D_STATE;
#pragma unroll
    for (int n = 0; n < D_STATE; ++n)
        hend[(hb + n) * D_INNER + d] = st[n];
    dsumb[(size_t)(b * NCH + ch) * D_INNER + d] = dsum;
}

__global__ void k_scan2(const float* __restrict__ aneg,
                        const float* __restrict__ hend,
                        const float* __restrict__ dsumb,
                        float* __restrict__ hinit)        // [B][NCH][16][768]
{
    int gid = blockIdx.x * blockDim.x + threadIdx.x;
    if (gid >= BB * D_INNER) return;
    int d = gid % D_INNER;
    int b = gid / D_INNER;
    float Ar[D_STATE], hc[D_STATE];
#pragma unroll
    for (int n = 0; n < D_STATE; ++n) { Ar[n] = aneg[d * D_STATE + n]; hc[n] = 0.f; }
    for (int ch = 0; ch < NCH; ++ch) {
        size_t hb = ((size_t)(b * NCH + ch)) * D_STATE;
#pragma unroll
        for (int n = 0; n < D_STATE; ++n)
            hinit[(hb + n) * D_INNER + d] = hc[n];
        float S = dsumb[(size_t)(b * NCH + ch) * D_INNER + d];
#pragma unroll
        for (int n = 0; n < D_STATE; ++n)
            hc[n] = hc[n] * __expf(S * Ar[n]) + hend[(hb + n) * D_INNER + d];
    }
}

__global__ void k_scan3(const float* __restrict__ delta,
                        const float* __restrict__ xdbl,
                        const float* __restrict__ aneg,
                        const float* __restrict__ hinit,
                        const float* __restrict__ ylocal,
                        const float* __restrict__ xz,     // z at col 768
                        unsigned short* __restrict__ y_bf)
{
    int gid = blockIdx.x * blockDim.x + threadIdx.x;
    if (gid >= BB * D_INNER * NCH) return;
    int d  = gid % D_INNER;
    int b  = (gid / D_INNER) % BB;
    int ch = gid / (D_INNER * BB);

    const int l0 = ch * CHL;
    const int l1 = (l0 + CHL < L_FULL) ? (l0 + CHL) : L_FULL;
    size_t base = (size_t)b * L_FULL;

    float Ar[D_STATE], hi[D_STATE];
    size_t hb = ((size_t)(b * NCH + ch)) * D_STATE;
#pragma unroll
    for (int n = 0; n < D_STATE; ++n) {
        Ar[n] = aneg[d * D_STATE + n];
        hi[n] = hinit[(hb + n) * D_INNER + d];
    }

    float dsum = 0.f;
    for (int l = l0; l < l1; ++l) {
        size_t row = base + l;
        if (l + 4 < l1)
            __builtin_prefetch(&ylocal[(row + 4) * D_INNER + d], 0, 1);
        float y = ylocal[row * D_INNER + d];
        if (ch != 0) {                       // chunk 0: h_init == 0
            float dl = delta[row * D_INNER + d];
            dsum += dl;
            const float* bc = xdbl + row * XP_N;
            float corr = 0.f;
#pragma unroll
            for (int n = 0; n < D_STATE; ++n)
                corr += bc[DT_RANK + D_STATE + n] * __expf(dsum * Ar[n]) * hi[n];
            y += corr;
        }
        float zv = xz[row * 1536 + D_INNER + d];
        float sg = zv / (1.f + __expf(-zv));
        y_bf[row * D_INNER + d] = f2bf(y * sg);
    }
}

// ---------------------------------------------------------------------------
// Batch-norm over (B,L): coalesced partials -> deterministic combine -> apply.
// ---------------------------------------------------------------------------
__global__ void k_bn_part(const float* __restrict__ h, float* __restrict__ part)
{
    int n   = threadIdx.x;                 // 384 threads: lane = channel
    int blk = blockIdx.x;                  // 0..63
    int r0 = blk * BN_RPB;
    int r1 = r0 + BN_RPB; if (r1 > M_FULL) r1 = M_FULL;
    float s = 0.f, s2 = 0.f;
    for (int r = r0; r < r1; ++r) {
        float v = h[(size_t)r * D_MODEL + n];
        s += v; s2 += v * v;
    }
    part[blk * D_MODEL + n] = s;
    part[BN_RB * D_MODEL + blk * D_MODEL + n] = s2;
}

__global__ void k_bn_stats(const float* __restrict__ part, float* __restrict__ stats)
{
    int n = blockIdx.x * blockDim.x + threadIdx.x;
    if (n >= D_MODEL) return;
    float s = 0.f, s2 = 0.f;
    for (int b = 0; b < BN_RB; ++b) {
        s  += part[b * D_MODEL + n];
        s2 += part[BN_RB * D_MODEL + b * D_MODEL + n];
    }
    float mu  = s * (1.f / M_FULL);
    float var = s2 * (1.f / M_FULL) - mu * mu;
    stats[n] = mu;
    stats[D_MODEL + n] = rsqrtf(var + EPSV);
}

__global__ void k_bn_apply(const float* __restrict__ h,
                           const float* __restrict__ stats,
                           const float* __restrict__ g,
                           const float* __restrict__ bta,
                           float* __restrict__ out)
{
    size_t idx = blockIdx.x * (size_t)blockDim.x + threadIdx.x;
    if (idx >= (size_t)M_FULL * D_MODEL) return;
    int n = (int)(idx % D_MODEL);
    out[idx] = (h[idx] - stats[n]) * stats[D_MODEL + n] * g[n] + bta[n];
}

// ---------------------------------------------------------------------------
// Host orchestration
// ---------------------------------------------------------------------------
extern "C" void kernel_launch(void* const* d_in, const int* in_sizes, int n_in,
                              void* d_out, int out_size, void* d_ws, size_t ws_size,
                              hipStream_t stream)
{
    const float* x      = (const float*)d_in[0];
    const float* norm_g = (const float*)d_in[4];
    const float* norm_b = (const float*)d_in[5];
    const float* W_red  = (const float*)d_in[6];
    const float* W_cls  = (const float*)d_in[7];
    const float* b_cls  = (const float*)d_in[8];
    const float* W_in   = (const float*)d_in[9];
    const float* conv_w = (const float*)d_in[10];
    const float* conv_b = (const float*)d_in[11];
    const float* W_xp   = (const float*)d_in[12];
    const float* W_dt   = (const float*)d_in[13];
    const float* b_dt   = (const float*)d_in[14];
    const float* A_log  = (const float*)d_in[15];
    const float* D_skp  = (const float*)d_in[16];
    const float* W_out  = (const float*)d_in[17];
    const float* bn_g   = (const float*)d_in[18];
    const float* bn_b   = (const float*)d_in[19];
    (void)in_sizes; (void)n_in; (void)out_size; (void)ws_size;

    // ---- workspace carve ----
    char* ws = (char*)d_ws;
    size_t off = 0;
    auto carve = [&](size_t bytes) -> char* {
        char* p = ws + off;
        off += (bytes + 255) & ~(size_t)255;
        return p;
    };
    float*          h_f32  = (float*)         carve((size_t)M_FULL * D_MODEL * 4);
    unsigned short* h_bf   = (unsigned short*)carve((size_t)M_FULL * D_MODEL * 2);
    unsigned short* m_bf   = (unsigned short*)carve((size_t)M_MERGE * D_INNER * 2);
    float*          m_out  = (float*)         carve((size_t)M_MERGE * D_MODEL * 4);
    float*          xz     = (float*)         carve((size_t)M_FULL * 1536 * 4);
    float*          xs     = (float*)         carve((size_t)M_FULL * D_INNER * 4);
    unsigned short* xs_bf  = (unsigned short*)carve((size_t)M_FULL * D_INNER * 2);
    float*          xdbl   = (float*)         carve((size_t)M_FULL * XP_N * 4);
    unsigned short* dt_bf  = (unsigned short*)carve((size_t)M_FULL * DT_PAD * 2);
    float*          delta  = (float*)         carve((size_t)M_FULL * D_INNER * 4);
    unsigned short* y_bf   = (unsigned short*)carve((size_t)M_FULL * D_INNER * 2);
    float*          ylocal = (float*)         carve((size_t)M_FULL * D_INNER * 4);
    float*          hend   = (float*)         carve((size_t)BB * NCH * D_STATE * D_INNER * 4);
    float*          hinit  = (float*)         carve((size_t)BB * NCH * D_STATE * D_INNER * 4);
    float*          dsumb  = (float*)         carve((size_t)BB * NCH * D_INNER * 4);
    unsigned short* wred_b = (unsigned short*)carve((size_t)D_MODEL * 768 * 2);
    unsigned short* win_b  = (unsigned short*)carve((size_t)2 * 1536 * D_MODEL * 2);
    unsigned short* wxp_b  = (unsigned short*)carve((size_t)2 * XP_N * D_INNER * 2);
    unsigned short* wdt_b  = (unsigned short*)carve((size_t)2 * D_INNER * DT_PAD * 2);
    unsigned short* wout_b = (unsigned short*)carve((size_t)2 * D_MODEL * D_INNER * 2);
    float*          aneg   = (float*)         carve((size_t)2 * D_INNER * D_STATE * 4);
    float*          bnpart = (float*)         carve((size_t)2 * BN_RB * D_MODEL * 4);
    float*          bnstat = (float*)         carve((size_t)2 * D_MODEL * 4);

    auto cvt = [&](const float* s, unsigned short* d, int n) {
        int blocks = (n + 255) / 256; if (blocks > 4096) blocks = 4096;
        k_f2bf<<<blocks, 256, 0, stream>>>(s, d, n);
    };
    auto gemm = [&](const unsigned short* A, const unsigned short* W, float* C,
                    int M, int N, int K, const float* bias,
                    unsigned short* Cbf, int mode) {
        dim3 grid((N + 16 * NT - 1) / (16 * NT), (M + 127) / 128);
        k_gemm_bf16<<<grid, 256, 0, stream>>>(A, W, C, M, N, K, bias, Cbf, mode);
    };

    // ---- weight prep (bf16) ----
    cvt(W_red, wred_b, D_MODEL * 768);
    cvt(W_in,  win_b,  2 * 1536 * D_MODEL);
    cvt(W_xp,  wxp_b,  2 * XP_N * D_INNER);
    cvt(W_out, wout_b, 2 * D_MODEL * D_INNER);
    k_wdt_pad<<<(2 * D_INNER * DT_PAD + 255) / 256, 256, 0, stream>>>(W_dt, wdt_b);
    k_aneg<<<(2 * D_INNER * D_STATE + 255) / 256, 256, 0, stream>>>(
        A_log, aneg, 2 * D_INNER * D_STATE);

    // ---- patch merge + LN + reduction GEMM + cls ----
    k_patch_merge_ln<<<(BB * 2048) / 8, 256, 0, stream>>>(x, norm_g, norm_b, m_bf);
    gemm(m_bf, wred_b, m_out, M_MERGE, D_MODEL, 768, nullptr, nullptr, 0);
    k_assemble_h<<<((size_t)M_FULL * D_MODEL + 255) / 256, 256, 0, stream>>>(
        m_out, x, W_cls, b_cls, h_f32, h_bf);

    // ---- Mamba layers ----
    const int scan_lanes = BB * D_INNER * NCH;
    for (int i = 0; i < 2; ++i) {
        gemm(h_bf, win_b + (size_t)i * 1536 * D_MODEL, xz,
             M_FULL, 1536, D_MODEL, nullptr, nullptr, 0);
        k_conv_silu<<<((size_t)M_FULL * D_INNER + 255) / 256, 256, 0, stream>>>(
            xz, conv_w + (size_t)i * D_INNER * 4, conv_b + (size_t)i * D_INNER,
            xs, xs_bf);
        gemm(xs_bf, wxp_b + (size_t)i * XP_N * D_INNER, xdbl,
             M_FULL, XP_N, D_INNER, nullptr, nullptr, 0);
        k_dt_pad<<<((size_t)M_FULL * DT_PAD + 255) / 256, 256, 0, stream>>>(
            xdbl, dt_bf);
        gemm(dt_bf, wdt_b + (size_t)i * D_INNER * DT_PAD, delta,
             M_FULL, D_INNER, DT_PAD, b_dt + (size_t)i * D_INNER, nullptr, 1);

        const float* an = aneg + (size_t)i * D_INNER * D_STATE;
        k_scan1<<<(scan_lanes + 255) / 256, 256, 0, stream>>>(
            xs, delta, xdbl, an, D_skp + (size_t)i * D_INNER,
            ylocal, hend, dsumb);
        k_scan2<<<(BB * D_INNER + 255) / 256, 256, 0, stream>>>(
            an, hend, dsumb, hinit);
        k_scan3<<<(scan_lanes + 255) / 256, 256, 0, stream>>>(
            delta, xdbl, an, hinit, ylocal, xz, y_bf);

        gemm(y_bf, wout_b + (size_t)i * D_MODEL * D_INNER, h_f32,
             M_FULL, D_MODEL, 768, nullptr, h_bf, 2);
    }

    // ---- final batch-norm ----
    k_bn_part<<<BN_RB, D_MODEL, 0, stream>>>(h_f32, bnpart);
    k_bn_stats<<<(D_MODEL + 127) / 128, 128, 0, stream>>>(bnpart, bnstat);
    k_bn_apply<<<((size_t)M_FULL * D_MODEL + 255) / 256, 256, 0, stream>>>(
        h_f32, bnstat, bn_g, bn_b, (float*)d_out);
}
